// FPCellLineMLPPredictor_16295105921296
// MI455X (gfx1250) — matrix operand order repaired and checked by reading, not compiled, not run on hardware
//
#include <hip/hip_runtime.h>
#include <hip/hip_bf16.h>

// ---------------------------------------------------------------------------
// Types for CDNA5 WMMA (wave32): V_WMMA_F32_16X16X32_BF16
// ---------------------------------------------------------------------------
typedef __attribute__((ext_vector_type(16))) __bf16 v16bf;
typedef __attribute__((ext_vector_type(8)))  __bf16 v8bf;
typedef __attribute__((ext_vector_type(8)))  float  v8f;

union AFrag { v16bf v; v8bf h[2]; };

// Force GLOBAL (addrspace 1) memory ops so the compiler never falls back to
// FLAT (which ticks both LOADcnt and DScnt on CDNA5).
__device__ __forceinline__ v8bf gload8(const __bf16* p) {
    return *(const __attribute__((address_space(1))) v8bf*)p;
}
__device__ __forceinline__ v16bf gload16(const __bf16* p) {
    return *(const __attribute__((address_space(1))) v16bf*)p;
}
__device__ __forceinline__ float gloadf(const float* p) {
    return *(const __attribute__((address_space(1))) float*)p;
}
__device__ __forceinline__ void gstore_bf(__bf16* p, __bf16 v) {
    *(__attribute__((address_space(1))) __bf16*)p = v;
}
__device__ __forceinline__ void gstore_f(float* p, float v) {
    *(__attribute__((address_space(1))) float*)p = v;
}

__device__ __forceinline__ __bf16 f2bf(float f) {
    // round-to-nearest-even f32 -> bf16
    unsigned u = __float_as_uint(f);
    unsigned r = (u + 0x7FFFu + ((u >> 16) & 1u)) >> 16;
    unsigned short s = (unsigned short)r;
    __bf16 b;
    __builtin_memcpy(&b, &s, 2);
    return b;
}

// One contiguous K-segment of the register-blocked WMMA GEMM.
// ap[s]: per-M-subtile A row base (already offset to this segment's start)
// bp[t]: per-N-subtile W column base (N-major); vkbase = segment's virtual-K start
__device__ __forceinline__ void wmma_seg(const __bf16* const* ap, int len,
                                         const __bf16* const* bp, int vkbase,
                                         int half, v8f acc[4][2]) {
    for (int kc = 0; kc < len; kc += 32) {
        AFrag a[4];
#pragma unroll
        for (int s = 0; s < 4; ++s) {
            a[s].h[0] = gload8(ap[s] + kc + 8 * half);       // K = kc + 8*half + [0..7]
            a[s].h[1] = gload8(ap[s] + kc + 16 + 8 * half);  // K = kc + 16 + 8*half + [0..7]
        }
        AFrag bf[2];
#pragma unroll
        for (int t = 0; t < 2; ++t) {
            bf[t].h[0] = gload8(bp[t] + vkbase + kc + 8 * half);       // K = kc + 8*half + [0..7]
            bf[t].h[1] = gload8(bp[t] + vkbase + kc + 16 + 8 * half);  // K = kc + 16 + 8*half + [0..7]
        }

#pragma unroll
        for (int s = 0; s < 4; ++s)
#pragma unroll
            for (int t = 0; t < 2; ++t)
                acc[s][t] = __builtin_amdgcn_wmma_f32_16x16x32_bf16(
                    false, a[s].v, false, bf[t].v, (short)0, acc[s][t], false, false);
    }
}

// ---------------------------------------------------------------------------
// Precompute kernels
// ---------------------------------------------------------------------------
__global__ void cvt_f32_bf16(const float* __restrict__ in, __bf16* __restrict__ out, int n) {
    int i = blockIdx.x * 256 + threadIdx.x;
    if (i < n) out[i] = f2bf(in[i]);
}

// W1t[n][vk] = bf16(W1[wrow(vk)][n]); vk in [0,2560), skipping attr rows 1280 & 2561
__global__ void build_W1t(const float* __restrict__ W1, __bf16* __restrict__ W1t) {
    int idx = blockIdx.x * 256 + threadIdx.x;            // 2048*2560
    if (idx >= 2048 * 2560) return;
    int n  = idx / 2560;
    int vk = idx - n * 2560;
    int k  = (vk < 1280) ? vk : vk + 1;
    W1t[(size_t)n * 2560 + vk] = f2bf(W1[(size_t)k * 2048 + n]);
}

// W2t[n][k] = bf16(W2[k][n]);  W2 is [2048][1024]
__global__ void build_W2t(const float* __restrict__ W2, __bf16* __restrict__ W2t) {
    int idx = blockIdx.x * 256 + threadIdx.x;            // 1024*2048
    if (idx >= 1024 * 2048) return;
    int n = idx / 2048;
    int k = idx - n * 2048;
    W2t[(size_t)n * 2048 + k] = f2bf(W2[(size_t)k * 1024 + n]);
}

// biasE[b][n] = b1[n] + attrs[b][0]*W1[1280][n] + attrs[b][1]*W1[2561][n]
__global__ void build_bias(const float* __restrict__ attrs, const float* __restrict__ W1,
                           const float* __restrict__ b1, float* __restrict__ biasE) {
    int idx = blockIdx.x * 256 + threadIdx.x;            // 2048*2048
    int b = idx >> 11;
    int n = idx & 2047;
    float v = b1[n] + attrs[2 * b] * W1[1280 * 2048 + n]
                    + attrs[2 * b + 1] * W1[2561u * 2048u + n];
    biasE[idx] = v;
}

// ---------------------------------------------------------------------------
// GEMM1: h1 = relu(batch @ W1 + biasE), batch gathered on the fly.
// M=2048, virtualK=2560 (segments: x[d1]1024 | h[d1]256 | x[d2]1024 | h[d2]256), N=2048
// Block: 128x128, 8 waves, wave tile 64x32 (4x2 WMMA 16x16x32 bf16 accumulators)
// K loop split into 4 straight-line segment loops (no per-iteration selection).
// ---------------------------------------------------------------------------
__global__ __launch_bounds__(256) void gemm1_wmma(
    const __bf16* __restrict__ x_bf,     // [4000][1024]
    const __bf16* __restrict__ h_bf,     // [4000][256]
    const __bf16* __restrict__ W1t,      // [2048][2560] (N-major, attr rows removed)
    const float*  __restrict__ biasE,    // [2048][2048]
    const int*    __restrict__ pair_idx, // [2048][2]
    __bf16*       __restrict__ h1)       // [2048][2048]
{
    const int lane = threadIdx.x & 31;
    const int wave = threadIdx.x >> 5;
    const int wm   = wave & 1;            // 2 wave-rows
    const int wn   = wave >> 1;           // 4 wave-cols
    const int half = lane >> 4;           // K-half select (A and B: +8)
    const int l16  = lane & 15;
    const int m0 = blockIdx.y * 128 + wm * 64;
    const int n0 = blockIdx.x * 128 + wn * 32;

    // Gathered A row pointers: A-fragment layout has M = lane%16
    const __bf16* xp1[4]; const __bf16* hp1[4];
    const __bf16* xp2[4]; const __bf16* hp2[4];
#pragma unroll
    for (int s = 0; s < 4; ++s) {
        int r  = m0 + s * 16 + l16;
        int i1 = pair_idx[2 * r];
        int i2 = pair_idx[2 * r + 1];
        xp1[s] = x_bf + (size_t)i1 * 1024;
        hp1[s] = h_bf + (size_t)i1 * 256;
        xp2[s] = x_bf + (size_t)i2 * 1024;
        hp2[s] = h_bf + (size_t)i2 * 256;
    }
    // B column pointers: B-fragment layout has N = lane%16, K = 8*(lane/16)+j (j<8), 16+8*(lane/16)+j-8 (j>=8)
    const __bf16* bp[2];
#pragma unroll
    for (int t = 0; t < 2; ++t) bp[t] = W1t + (size_t)(n0 + t * 16 + l16) * 2560;

    v8f acc[4][2];
#pragma unroll
    for (int s = 0; s < 4; ++s)
#pragma unroll
        for (int t = 0; t < 2; ++t) acc[s][t] = v8f{0, 0, 0, 0, 0, 0, 0, 0};

    // Virtual-K segments (each 32-aligned): [0,1024) x[d1] | [1024,1280) h[d1]
    //                                      | [1280,2304) x[d2] | [2304,2560) h[d2]
    wmma_seg(xp1, 1024, bp, 0,    half, acc);
    wmma_seg(hp1, 256,  bp, 1024, half, acc);
    wmma_seg(xp2, 1024, bp, 1280, half, acc);
    wmma_seg(hp2, 256,  bp, 2304, half, acc);

    // Epilogue: C/D layout M = r + 8*half, N = lane%16
#pragma unroll
    for (int s = 0; s < 4; ++s)
#pragma unroll
        for (int t = 0; t < 2; ++t)
#pragma unroll
            for (int r = 0; r < 8; ++r) {
                int row = m0 + s * 16 + r + 8 * half;
                int col = n0 + t * 16 + l16;
                float v = acc[s][t][r] + gloadf(biasE + (size_t)row * 2048 + col);
                gstore_bf(h1 + (size_t)row * 2048 + col, f2bf(fmaxf(v, 0.0f)));
            }
}

// ---------------------------------------------------------------------------
// GEMM2: h2 = relu(h1 @ W2 + b2). M=2048, K=2048, N=1024. Same tiling.
// ---------------------------------------------------------------------------
__global__ __launch_bounds__(256) void gemm2_wmma(
    const __bf16* __restrict__ h1,   // [2048][2048] bf16
    const __bf16* __restrict__ W2t,  // [1024][2048] N-major
    const float*  __restrict__ b2,   // [1024]
    float*        __restrict__ h2)   // [2048][1024] f32
{
    const int lane = threadIdx.x & 31;
    const int wave = threadIdx.x >> 5;
    const int wm   = wave & 1;
    const int wn   = wave >> 1;
    const int half = lane >> 4;
    const int l16  = lane & 15;
    const int m0 = blockIdx.y * 128 + wm * 64;
    const int n0 = blockIdx.x * 128 + wn * 32;

    const __bf16* ap[4];
#pragma unroll
    for (int s = 0; s < 4; ++s) ap[s] = h1 + (size_t)(m0 + s * 16 + l16) * 2048;
    const __bf16* bp[2];
#pragma unroll
    for (int t = 0; t < 2; ++t) bp[t] = W2t + (size_t)(n0 + t * 16 + l16) * 2048;

    v8f acc[4][2];
#pragma unroll
    for (int s = 0; s < 4; ++s)
#pragma unroll
        for (int t = 0; t < 2; ++t) acc[s][t] = v8f{0, 0, 0, 0, 0, 0, 0, 0};

    wmma_seg(ap, 2048, bp, 0, half, acc);

#pragma unroll
    for (int s = 0; s < 4; ++s)
#pragma unroll
        for (int t = 0; t < 2; ++t)
#pragma unroll
            for (int r = 0; r < 8; ++r) {
                int row = m0 + s * 16 + r + 8 * half;
                int col = n0 + t * 16 + l16;
                float v = acc[s][t][r] + gloadf(b2 + col);
                gstore_f(h2 + (size_t)row * 1024 + col, fmaxf(v, 0.0f));
            }
}

// ---------------------------------------------------------------------------
// Cell-line heads: per-sample 128x1024 matvec + relu + 128-dot head.
// One block (8 waves) per sample; h2[b] staged in LDS; shfl_xor reductions.
// ---------------------------------------------------------------------------
__global__ __launch_bounds__(256) void cell_head(
    const float* __restrict__ h2,   // [2048][1024]
    const float* __restrict__ CW1,  // [64][128][1024]
    const float* __restrict__ CO1,  // [64][128]
    const float* __restrict__ CW2,  // [64][1][128]
    const float* __restrict__ CO2,  // [64][1]
    const int*   __restrict__ cell_lines,
    float*       __restrict__ out)  // [2048]
{
    __shared__ float sh[1024];
    __shared__ float h3[128];
    const int b  = blockIdx.x;
    const int cl = cell_lines[b];
    const float* hv = h2 + (size_t)b * 1024;
    for (int i = threadIdx.x; i < 1024; i += 256) sh[i] = gloadf(hv + i);
    __syncthreads();

    const int lane = threadIdx.x & 31;
    const int wave = threadIdx.x >> 5;
    const float* Wc = CW1 + (size_t)cl * 128 * 1024;
    for (int o = wave; o < 128; o += 8) {
        const float* wr = Wc + (size_t)o * 1024;
        float acc = 0.0f;
        for (int i = lane; i < 1024; i += 32) acc = fmaf(gloadf(wr + i), sh[i], acc);
        for (int off = 16; off > 0; off >>= 1) acc += __shfl_xor(acc, off, 32);
        if (lane == 0) h3[o] = fmaxf(acc + CO1[cl * 128 + o], 0.0f);
    }
    __syncthreads();
    if (wave == 0) {
        const float* w2 = CW2 + (size_t)cl * 128;
        float acc = 0.0f;
        for (int o = lane; o < 128; o += 32) acc = fmaf(gloadf(w2 + o), h3[o], acc);
        for (int off = 16; off > 0; off >>= 1) acc += __shfl_xor(acc, off, 32);
        if (lane == 0) gstore_f(out + b, acc + CO2[cl]);
    }
}

// ---------------------------------------------------------------------------
// Launch
// ---------------------------------------------------------------------------
extern "C" void kernel_launch(void* const* d_in, const int* in_sizes, int n_in,
                              void* d_out, int out_size, void* d_ws, size_t ws_size,
                              hipStream_t stream) {
    const float* x_drugs    = (const float*)d_in[0];   // 4000*1024
    const float* h_drug     = (const float*)d_in[1];   // 4000*256
    const float* attrs      = (const float*)d_in[2];   // 2048*2
    const float* W1         = (const float*)d_in[3];   // 2562*2048
    const float* b1         = (const float*)d_in[4];   // 2048
    const float* W2         = (const float*)d_in[5];   // 2048*1024
    const float* b2         = (const float*)d_in[6];   // 1024
    const float* CW1        = (const float*)d_in[7];   // 64*128*1024
    const float* CO1        = (const float*)d_in[8];   // 64*128
    const float* CW2        = (const float*)d_in[9];   // 64*1*128
    const float* CO2        = (const float*)d_in[10];  // 64*1
    const int*   pair_idx   = (const int*)d_in[11];    // 2048*2
    const int*   cell_lines = (const int*)d_in[12];    // 2048
    float* out = (float*)d_out;

    char* ws = (char*)d_ws;
    __bf16* x_bf  = (__bf16*)(ws + 0);          //  8,192,000 B : [4000][1024]
    __bf16* h_bf  = (__bf16*)(ws + 8192000);    //  2,048,000 B : [4000][256]
    __bf16* W1t   = (__bf16*)(ws + 10240000);   // 10,485,760 B : [2048][2560]
    __bf16* W2t   = (__bf16*)(ws + 20725760);   //  4,194,304 B : [1024][2048]
    float*  biasE = (float* )(ws + 24920064);   // 16,777,216 B : [2048][2048]
    __bf16* h1    = (__bf16*)(ws + 41697280);   //  8,388,608 B : [2048][2048]
    float*  h2    = (float* )(ws + 50085888);   //  8,388,608 B : [2048][1024]
                                                // total 58,474,496 B

    cvt_f32_bf16<<<(4096000 + 255) / 256, 256, 0, stream>>>(x_drugs, x_bf, 4096000);
    cvt_f32_bf16<<<(1024000 + 255) / 256, 256, 0, stream>>>(h_drug, h_bf, 1024000);
    build_W1t<<<(2048 * 2560) / 256, 256, 0, stream>>>(W1, W1t);
    build_W2t<<<(1024 * 2048) / 256, 256, 0, stream>>>(W2, W2t);
    build_bias<<<(2048 * 2048) / 256, 256, 0, stream>>>(attrs, W1, b1, biasE);

    gemm1_wmma<<<dim3(16, 16), 256, 0, stream>>>(x_bf, h_bf, W1t, biasE, pair_idx, h1);
    gemm2_wmma<<<dim3(8, 16), 256, 0, stream>>>(h1, W2t, b2, h2);
    cell_head<<<2048, 256, 0, stream>>>(h2, CW1, CO1, CW2, CO2, cell_lines, out);
}
